// model_S0_56521769616066
// MI455X (gfx1250) — matrix operand order repaired and checked by reading, not compiled, not run on hardware
//
#include <hip/hip_runtime.h>
#include <hip/hip_bf16.h>

#define N_NODES 65536
#define N_EDGES 524288
#define IN_CH   128
#define HEADS   4
#define HID     16
#define HC      64
#define EDGE_F  8
#define N_GRAPHS 512

typedef __attribute__((ext_vector_type(16))) __bf16 v16bf;
typedef __attribute__((ext_vector_type(8)))  float  v8f;

// ---------- float <-> order-preserving uint (for atomic segment max) ----------
__device__ __forceinline__ unsigned orderF(float f) {
  unsigned u = __float_as_uint(f);
  return (u & 0x80000000u) ? ~u : (u | 0x80000000u);
}
__device__ __forceinline__ float unorderF(unsigned o) {
  unsigned u = (o & 0x80000000u) ? (o & 0x7FFFFFFFu) : ~o;
  return __uint_as_float(u);
}

// ---------------------------- utility: zero buffer ----------------------------
__global__ void zero_kernel(float* __restrict__ p, int n) {
  int t = blockIdx.x * blockDim.x + threadIdx.x;
  if (t < n) p[t] = 0.0f;
}

// ------------- pack W[K x 64] (f32, row-major) into WMMA-B bf16 layout --------
// out layout: [nt(4)][kc(K/32)][lane(32)][j(16)], lane holds col=(lane&15)+nt*16,
// j enumerates K = kc*32 + (lane<16 ? 0 : 8) + j + (j>=8 ? 8 : 0).
__global__ void pack_weight(const float* __restrict__ W, int K, __bf16* __restrict__ out) {
  int t = blockIdx.x * blockDim.x + threadIdx.x;
  int lane = t & 31;
  int wid  = t >> 5;
  int KC = K >> 5;
  if (wid >= 4 * KC) return;
  int nt = wid / KC, kc = wid - nt * KC;
  int col = (lane & 15) + nt * 16;
  int kbase = kc * 32 + ((lane & 16) ? 8 : 0);
  __bf16* dst = out + ((size_t)wid * 32 + lane) * 16;
#pragma unroll
  for (int j = 0; j < 16; ++j)
    dst[j] = (__bf16)W[(size_t)(kbase + j + (j & 8)) * HC + col];
}

// ---------------- fused node projection: q,k,v and skip (h init) --------------
// Each block: 128 rows staged in LDS (bf16), 8 waves x 16-row WMMA tiles.
template <int K, bool RELU>
__global__ __launch_bounds__(256) void proj_kernel(
    const float* __restrict__ x,
    const __bf16* __restrict__ pwq, const __bf16* __restrict__ pwk,
    const __bf16* __restrict__ pwv, const __bf16* __restrict__ pws,
    const float* __restrict__ bq, const float* __restrict__ bk,
    const float* __restrict__ bv, const float* __restrict__ bs,
    float* __restrict__ oq, float* __restrict__ ok,
    float* __restrict__ ov, float* __restrict__ oh)
{
  constexpr int KC  = K / 32;
  constexpr int LDX = K + 2;              // pad 2 bf16 (4B) -> rotate banks per row
  __shared__ __bf16 xs[128 * LDX];

  const int tid = threadIdx.x;
  const int rowbase = blockIdx.x * 128;

  // cooperative load + f32->bf16 convert (+ optional ReLU)
  for (int idx = tid; idx < 128 * K; idx += 256) {
    int r = idx / K, c = idx - r * K;
    float f = x[(size_t)(rowbase + r) * K + c];
    if (RELU) f = fmaxf(f, 0.0f);
    xs[r * LDX + c] = (__bf16)f;
  }
  __syncthreads();

  const int lane = tid & 31;
  const int wave = tid >> 5;
  const int mrow = wave * 16 + (lane & 15);
  const int kb0  = (lane & 16) ? 8 : 0;

  // Build A fragments per ISA 16-bit A layout: elems[0..7]=K kb0+j, [8..15]=K 16+kb0+j
  v16bf A[KC];
#pragma unroll
  for (int kc = 0; kc < KC; ++kc) {
    const __bf16* p = &xs[mrow * LDX + kc * 32 + kb0];
#pragma unroll
    for (int j = 0; j < 8; ++j) { A[kc][j] = p[j]; A[kc][8 + j] = p[16 + j]; }
  }

  const __bf16* pw[4]  = {pwq, pwk, pwv, pws};
  const float* bias[4] = {bq, bk, bv, bs};
  float* outp[4]       = {oq, ok, ov, oh};
  const int rb = rowbase + wave * 16 + ((lane & 16) ? 8 : 0);

#pragma unroll
  for (int w = 0; w < 4; ++w) {
#pragma unroll
    for (int nt = 0; nt < 4; ++nt) {
      float bval = bias[w][nt * 16 + (lane & 15)];
      v8f acc;
#pragma unroll
      for (int r = 0; r < 8; ++r) acc[r] = bval;
#pragma unroll
      for (int kc = 0; kc < KC; ++kc) {
        v16bf B = *(const v16bf*)(pw[w] + ((size_t)((nt * KC + kc) * 32 + lane)) * 16);
        acc = __builtin_amdgcn_wmma_f32_16x16x32_bf16(
            false, A[kc], false, B, (short)0, acc, false, false);
      }
      int col = nt * 16 + (lane & 15);
      float* o = outp[w];
#pragma unroll
      for (int r = 0; r < 8; ++r)
        o[(size_t)(rb + r) * HC + col] = acc[r];
    }
  }
}

// ---------------- pass 1: per-(edge,head) attention logit + segment max -------
__global__ void edge_logits(const float* __restrict__ q, const float* __restrict__ k,
                            const float* __restrict__ ea,
                            const float* __restrict__ We, const float* __restrict__ be,
                            const int* __restrict__ src, const int* __restrict__ dst,
                            float* __restrict__ logits, unsigned* __restrict__ mord, int n)
{
  int t = blockIdx.x * blockDim.x + threadIdx.x;
  if (t >= n) return;
  int e = t >> 2, h = t & 3;
  int s = src[e], d = dst[e];
  const float* qp = q + (size_t)d * HC + h * HID;
  const float* kp = k + (size_t)s * HC + h * HID;
  float ef[EDGE_F];
#pragma unroll
  for (int f = 0; f < EDGE_F; ++f) ef[f] = ea[(size_t)e * EDGE_F + f];
  float acc = 0.0f;
#pragma unroll
  for (int c = 0; c < HID; ++c) {
    int col = h * HID + c;
    float ec = be[col];
#pragma unroll
    for (int f = 0; f < EDGE_F; ++f) ec = fmaf(ef[f], We[f * HC + col], ec);
    acc = fmaf(qp[c], kp[c] + ec, acc);
  }
  acc *= 0.25f;  // 1/sqrt(HID)
  logits[t] = acc;
  atomicMax(&mord[(size_t)d * HEADS + h], orderF(acc));
}

// ---------------- pass 2: exp(logit - max) + segment denom --------------------
__global__ void edge_softmax(const float* __restrict__ logits, float* __restrict__ alpha,
                             const unsigned* __restrict__ mord,
                             const int* __restrict__ dst, float* __restrict__ denom, int n)
{
  int t = blockIdx.x * blockDim.x + threadIdx.x;
  if (t >= n) return;
  int e = t >> 2, h = t & 3;
  int d = dst[e];
  float m = unorderF(mord[(size_t)d * HEADS + h]);
  float a = __expf(logits[t] - m);
  alpha[t] = a;
  atomicAdd(&denom[(size_t)d * HEADS + h], a);
}

// ---------------- pass 3: weighted message scatter-add ------------------------
__global__ void edge_message(const float* __restrict__ v, const float* __restrict__ ea,
                             const float* __restrict__ We, const float* __restrict__ be,
                             const int* __restrict__ src, const int* __restrict__ dst,
                             const float* __restrict__ alpha, const float* __restrict__ denom,
                             float* __restrict__ hout, int n)
{
  int t = blockIdx.x * blockDim.x + threadIdx.x;
  if (t >= n) return;
  int e = t >> 6, c = t & 63, h = c >> 4;
  int s = src[e], d = dst[e];
  float ec = be[c];
#pragma unroll
  for (int f = 0; f < EDGE_F; ++f) ec = fmaf(ea[(size_t)e * EDGE_F + f], We[f * HC + c], ec);
  float a  = alpha[(size_t)e * HEADS + h];
  float dn = denom[(size_t)d * HEADS + h];
  float al = a / fmaxf(dn, 1e-16f);
  float msg = (v[(size_t)s * HC + c] + ec) * al;
  atomicAdd(&hout[(size_t)d * HC + c], msg);
}

// ---------------- pooling ------------------------------------------------------
__global__ void pool_sum(const float* __restrict__ h, const int* __restrict__ batch,
                         float* __restrict__ out, float* __restrict__ counts, int n)
{
  int t = blockIdx.x * blockDim.x + threadIdx.x;
  if (t >= n) return;
  int node = t >> 6, c = t & 63;
  int g = batch[node];
  atomicAdd(&out[(size_t)g * HC + c], h[(size_t)node * HC + c]);
  if (c == 0) atomicAdd(&counts[g], 1.0f);
}

__global__ void pool_norm(float* __restrict__ out, const float* __restrict__ counts, int n)
{
  int t = blockIdx.x * blockDim.x + threadIdx.x;
  if (t >= n) return;
  out[t] /= fmaxf(counts[t >> 6], 1.0f);
}

// ---------------- workspace layout (bytes) ------------------------------------
static const size_t NODE_BUF = (size_t)N_NODES * HC * 4;           // 16 MB
static const size_t OFF_Q     = 0;
static const size_t OFF_K     = NODE_BUF;
static const size_t OFF_V     = 2 * NODE_BUF;
static const size_t OFF_H1    = 3 * NODE_BUF;
static const size_t OFF_H2    = 4 * NODE_BUF;
static const size_t OFF_LOG   = 5 * NODE_BUF;                      // E*H*4 = 8 MB
static const size_t OFF_M     = OFF_LOG + (size_t)N_EDGES * HEADS * 4;
static const size_t OFF_DEN   = OFF_M + (size_t)N_NODES * HEADS * 4;
static const size_t OFF_CNT   = OFF_DEN + (size_t)N_NODES * HEADS * 4;
static const size_t OFF_PW    = OFF_CNT + 4096;
static const size_t PW1_SZ    = (size_t)IN_CH * HC * 2;            // 16 KB each
static const size_t PW2_SZ    = (size_t)HC * HC * 2;               // 8 KB each

extern "C" void kernel_launch(void* const* d_in, const int* in_sizes, int n_in,
                              void* d_out, int out_size, void* d_ws, size_t ws_size,
                              hipStream_t stream) {
  const float* x     = (const float*)d_in[0];
  const float* ea    = (const float*)d_in[1];
  const int*   ei    = (const int*)d_in[2];
  const int*   batch = (const int*)d_in[3];
  const float* Wq1 = (const float*)d_in[4];
  const float* Wk1 = (const float*)d_in[5];
  const float* Wv1 = (const float*)d_in[6];
  const float* We1 = (const float*)d_in[7];
  const float* Ws1 = (const float*)d_in[8];
  const float* Wq2 = (const float*)d_in[9];
  const float* Wk2 = (const float*)d_in[10];
  const float* Wv2 = (const float*)d_in[11];
  const float* We2 = (const float*)d_in[12];
  const float* Ws2 = (const float*)d_in[13];
  const float* bq1 = (const float*)d_in[14];
  const float* bk1 = (const float*)d_in[15];
  const float* bv1 = (const float*)d_in[16];
  const float* be1 = (const float*)d_in[17];
  const float* bs1 = (const float*)d_in[18];
  const float* bq2 = (const float*)d_in[19];
  const float* bk2 = (const float*)d_in[20];
  const float* bv2 = (const float*)d_in[21];
  const float* be2 = (const float*)d_in[22];
  const float* bs2 = (const float*)d_in[23];

  const int* src = ei;
  const int* dst = ei + N_EDGES;

  char* ws = (char*)d_ws;
  float*    q      = (float*)(ws + OFF_Q);
  float*    k      = (float*)(ws + OFF_K);
  float*    v      = (float*)(ws + OFF_V);
  float*    h1     = (float*)(ws + OFF_H1);
  float*    h2     = (float*)(ws + OFF_H2);
  float*    logits = (float*)(ws + OFF_LOG);
  unsigned* mord   = (unsigned*)(ws + OFF_M);
  float*    denom  = (float*)(ws + OFF_DEN);
  float*    counts = (float*)(ws + OFF_CNT);
  __bf16* pq1 = (__bf16*)(ws + OFF_PW);
  __bf16* pk1 = (__bf16*)(ws + OFF_PW + 1 * PW1_SZ);
  __bf16* pv1 = (__bf16*)(ws + OFF_PW + 2 * PW1_SZ);
  __bf16* ps1 = (__bf16*)(ws + OFF_PW + 3 * PW1_SZ);
  __bf16* pq2 = (__bf16*)(ws + OFF_PW + 4 * PW1_SZ);
  __bf16* pk2 = (__bf16*)(ws + OFF_PW + 4 * PW1_SZ + 1 * PW2_SZ);
  __bf16* pv2 = (__bf16*)(ws + OFF_PW + 4 * PW1_SZ + 2 * PW2_SZ);
  __bf16* ps2 = (__bf16*)(ws + OFF_PW + 4 * PW1_SZ + 3 * PW2_SZ);
  float* out = (float*)d_out;

  // ---- pack weights to WMMA-B bf16 layout (tiny; once per call) ----
  pack_weight<<<1, 512, 0, stream>>>(Wq1, IN_CH, pq1);
  pack_weight<<<1, 512, 0, stream>>>(Wk1, IN_CH, pk1);
  pack_weight<<<1, 512, 0, stream>>>(Wv1, IN_CH, pv1);
  pack_weight<<<1, 512, 0, stream>>>(Ws1, IN_CH, ps1);
  pack_weight<<<1, 256, 0, stream>>>(Wq2, HC, pq2);
  pack_weight<<<1, 256, 0, stream>>>(Wk2, HC, pk2);
  pack_weight<<<1, 256, 0, stream>>>(Wv2, HC, pv2);
  pack_weight<<<1, 256, 0, stream>>>(Ws2, HC, ps2);

  const int NH = N_NODES * HEADS;     // 262144
  const int EH = N_EDGES * HEADS;     // 2097152
  const int EC = N_EDGES * HC;        // 33554432
  const int NC = N_NODES * HC;        // 4194304
  const int GC = N_GRAPHS * HC;       // 32768

  // ---- init scratch/output ----
  zero_kernel<<<GC / 256, 256, 0, stream>>>(out, GC);
  zero_kernel<<<2, 256, 0, stream>>>(counts, N_GRAPHS);
  zero_kernel<<<NH / 256, 256, 0, stream>>>((float*)mord, NH);  // ordered-uint 0 == -inf-ish floor
  zero_kernel<<<NH / 256, 256, 0, stream>>>(denom, NH);

  // ---- layer 1 ----
  proj_kernel<IN_CH, false><<<N_NODES / 128, 256, 0, stream>>>(
      x, pq1, pk1, pv1, ps1, bq1, bk1, bv1, bs1, q, k, v, h1);
  edge_logits <<<EH / 256, 256, 0, stream>>>(q, k, ea, We1, be1, src, dst, logits, mord, EH);
  edge_softmax<<<EH / 256, 256, 0, stream>>>(logits, logits, mord, dst, denom, EH);
  edge_message<<<EC / 256, 256, 0, stream>>>(v, ea, We1, be1, src, dst, logits, denom, h1, EC);

  // ---- re-init softmax scratch for layer 2 ----
  zero_kernel<<<NH / 256, 256, 0, stream>>>((float*)mord, NH);
  zero_kernel<<<NH / 256, 256, 0, stream>>>(denom, NH);

  // ---- layer 2 (ReLU fused into projection load; skip uses ReLU'd input too) ----
  proj_kernel<HC, true><<<N_NODES / 128, 256, 0, stream>>>(
      h1, pq2, pk2, pv2, ps2, bq2, bk2, bv2, bs2, q, k, v, h2);
  edge_logits <<<EH / 256, 256, 0, stream>>>(q, k, ea, We2, be2, src, dst, logits, mord, EH);
  edge_softmax<<<EH / 256, 256, 0, stream>>>(logits, logits, mord, dst, denom, EH);
  edge_message<<<EC / 256, 256, 0, stream>>>(v, ea, We2, be2, src, dst, logits, denom, h2, EC);

  // ---- mean pooling ----
  pool_sum <<<NC / 256, 256, 0, stream>>>(h2, batch, out, counts, NC);
  pool_norm<<<GC / 256, 256, 0, stream>>>(out, counts, GC);
}
